// AttentionalLSTM_15831249453746
// MI455X (gfx1250) — matrix operand order repaired and checked by reading, not compiled, not run on hardware
//
#include <hip/hip_runtime.h>
#include <hip/hip_bf16.h>
#include <math.h>

// ---- problem dims ----
#define B_  64
#define T_  128
#define S_  256
#define I_  768
#define H_  768
#define G_  3072   // 4*H

typedef __attribute__((ext_vector_type(16))) __bf16 v16bf;
typedef __attribute__((ext_vector_type(8)))  __bf16 v8bf;
typedef __attribute__((ext_vector_type(8)))  float  v8f;

__device__ __forceinline__ __bf16 f2bf(float f) {
  union { float f; unsigned u; } a; a.f = f;
  unsigned r = a.u + 0x7FFFu + ((a.u >> 16) & 1u);   // round-to-nearest-even
  union { unsigned short s; __bf16 b; } o; o.s = (unsigned short)(r >> 16);
  return o.b;
}

__device__ __forceinline__ float sigm(float x) { return 1.0f / (1.0f + __expf(-x)); }

// A-matrix bf16 16x32 tile in contiguous k order (same as B), loaded straight from a bf16 row:
// lane half 0 holds K = k0+[0..15]; half 1 holds K = k0+[16..31]
__device__ __forceinline__ v16bf load_a_bf(const __bf16* __restrict__ arow, int k0, int half) {
  v8bf lo = *(const v8bf*)(arow + k0 + 16 * half);
  v8bf hi = *(const v8bf*)(arow + k0 + 16 * half + 8);
  return __builtin_shufflevector(lo, hi, 0, 1, 2, 3, 4, 5, 6, 7,
                                 8, 9, 10, 11, 12, 13, 14, 15);
}

// ---------------- fp32 -> bf16 conversion (weights + activations, done once) ----------------
__global__ __launch_bounds__(256) void cvt_bf16(const float* __restrict__ in,
                                                __bf16* __restrict__ outp, int n) {
  int i = blockIdx.x * blockDim.x + threadIdx.x;
  if (i < n) outp[i] = f2bf(in[i]);
}

// ------- generic C[M,N] = A[M,K] @ W[N,K]^T + bias1 (+bias2); 16x32 output per wave -------
__global__ __launch_bounds__(256) void gemm_bf_2n(const __bf16* __restrict__ A, int lda,
                                                  const __bf16* __restrict__ W,
                                                  const float* __restrict__ bias1,
                                                  const float* __restrict__ bias2,
                                                  float* __restrict__ C,
                                                  int M, int N, int K) {
  int wave = threadIdx.x >> 5;
  int tile = blockIdx.x * (blockDim.x >> 5) + wave;
  int tN = N >> 5, tM = M >> 4;
  if (tile >= tM * tN) return;
  int m0 = (tile / tN) << 4;
  int n0 = (tile % tN) << 5;
  int lane = threadIdx.x & 31;
  int l = lane & 15, half = lane >> 4;
  const __bf16* arow = A + (size_t)(m0 + l) * lda;
  const __bf16* w0 = W + (size_t)(n0 + l) * K;
  const __bf16* w1 = W + (size_t)(n0 + 16 + l) * K;
  v8f acc0 = {0.f, 0.f, 0.f, 0.f, 0.f, 0.f, 0.f, 0.f};
  v8f acc1 = acc0;
  for (int k0 = 0; k0 < K; k0 += 32) {
    __builtin_prefetch((const void*)(w0 + k0 + 256), 0, 1);
    __builtin_prefetch((const void*)(w1 + k0 + 256), 0, 1);
    v16bf av = load_a_bf(arow, k0, half);                 // reused across 2 B tiles
    v16bf b0 = *(const v16bf*)(w0 + k0 + 16 * half);      // 16 contiguous K, 32B aligned
    v16bf b1 = *(const v16bf*)(w1 + k0 + 16 * half);
    acc0 = __builtin_amdgcn_wmma_f32_16x16x32_bf16(false, av, false, b0, (short)0, acc0,
                                                   false, false);
    acc1 = __builtin_amdgcn_wmma_f32_16x16x32_bf16(false, av, false, b1, (short)0, acc1,
                                                   false, false);
  }
  int na = n0 + l, nb = n0 + 16 + l;
  float ba = bias1 ? bias1[na] : 0.f;
  float bb = bias1 ? bias1[nb] : 0.f;
  if (bias2) { ba += bias2[na]; bb += bias2[nb]; }
  for (int r = 0; r < 8; ++r) {
    int m = m0 + r + 8 * half;
    C[(size_t)m * N + na] = acc0[r] + ba;
    C[(size_t)m * N + nb] = acc1[r] + bb;
  }
}

// ---------------- fused LSTM cell: gates_h = h @ W_hh^T, 4 gate tiles / wave ----------------
__global__ __launch_bounds__(256) void lstm_step(const __bf16* __restrict__ h_bf,    // [B,H] bf16
                                                 const __bf16* __restrict__ Whh,     // [4H,H] bf16
                                                 const float* __restrict__ gates_x,  // [B*T,4H]
                                                 float* __restrict__ c_state,        // [B,H] in/out
                                                 __bf16* __restrict__ cat_bf,        // [B,2H]; hy -> +H
                                                 float* __restrict__ raw_out,        // [T,B,H]
                                                 int t) {
  int wave = threadIdx.x >> 5;
  int tile = blockIdx.x * (blockDim.x >> 5) + wave;
  const int tN = H_ >> 4;                       // 48
  if (tile >= (B_ / 16) * tN) return;           // 192 tiles
  int m0 = (tile / tN) << 4;
  int n0 = (tile % tN) << 4;
  int lane = threadIdx.x & 31;
  int l = lane & 15, half = lane >> 4;
  const __bf16* arow = h_bf + (size_t)(m0 + l) * H_;
  const __bf16* w0 = Whh + (size_t)(0 * H_ + n0 + l) * H_;
  const __bf16* w1 = Whh + (size_t)(1 * H_ + n0 + l) * H_;
  const __bf16* w2 = Whh + (size_t)(2 * H_ + n0 + l) * H_;
  const __bf16* w3 = Whh + (size_t)(3 * H_ + n0 + l) * H_;
  v8f ai = {0.f, 0.f, 0.f, 0.f, 0.f, 0.f, 0.f, 0.f};
  v8f af = ai, ag = ai, ao = ai;
  for (int k0 = 0; k0 < H_; k0 += 32) {
    v16bf av = load_a_bf(arow, k0, half);       // A reused across the 4 gate tiles
    v16bf b0 = *(const v16bf*)(w0 + k0 + 16 * half);
    v16bf b1 = *(const v16bf*)(w1 + k0 + 16 * half);
    v16bf b2 = *(const v16bf*)(w2 + k0 + 16 * half);
    v16bf b3 = *(const v16bf*)(w3 + k0 + 16 * half);
    ai = __builtin_amdgcn_wmma_f32_16x16x32_bf16(false, av, false, b0, (short)0, ai, false, false);
    af = __builtin_amdgcn_wmma_f32_16x16x32_bf16(false, av, false, b1, (short)0, af, false, false);
    ag = __builtin_amdgcn_wmma_f32_16x16x32_bf16(false, av, false, b2, (short)0, ag, false, false);
    ao = __builtin_amdgcn_wmma_f32_16x16x32_bf16(false, av, false, b3, (short)0, ao, false, false);
  }
  int n = n0 + l;
  for (int r = 0; r < 8; ++r) {
    int b = m0 + r + 8 * half;
    const float* gx = gates_x + ((size_t)b * T_ + t) * G_;   // biases already folded in
    float ig = ai[r] + gx[n];
    float fg = af[r] + gx[H_ + n];
    float gg = ag[r] + gx[2 * H_ + n];
    float og = ao[r] + gx[3 * H_ + n];
    size_t ci = (size_t)b * H_ + n;
    float c  = c_state[ci];
    float cy = sigm(fg) * c + sigm(ig) * tanhf(gg);
    float hy = sigm(og) * tanhf(cy);
    c_state[ci] = cy;                                        // in-place: unique owner
    cat_bf[(size_t)b * (2 * H_) + H_ + n] = f2bf(hy);        // hy half of concat (bf16)
    raw_out[((size_t)t * B_ + b) * H_ + n] = hy;             // fp32 output
  }
}

// ---------------- attention: scores -> softmax -> context, one block per batch ----------------
__global__ __launch_bounds__(256) void attention_step(const float* __restrict__ att_keys, // [B*S,H]
                                                      const float* __restrict__ qbuf,     // [B,H]
                                                      const unsigned char* __restrict__ mask, // [B,S]
                                                      __bf16* __restrict__ cat_bf,   // ctx -> [b,0..H)
                                                      float* __restrict__ attn_out,  // [T,B,S]
                                                      float* __restrict__ ctx_out,   // [T,B,H]
                                                      int t) {
  __shared__ float q_s[H_];
  __shared__ float p_s[S_];
  __shared__ float red[256];
  int b = blockIdx.x;
  int tid = threadIdx.x;
  for (int i = tid; i < H_; i += 256) q_s[i] = qbuf[(size_t)b * H_ + i];
  __syncthreads();
  int s = tid;                                   // 256 threads == S
  const float* kr = att_keys + ((size_t)b * S_ + s) * H_;
  float sc = 0.f;
  for (int k = 0; k < H_; k += 4) {
    float4 kv = *(const float4*)(kr + k);
    sc += kv.x * q_s[k] + kv.y * q_s[k + 1] + kv.z * q_s[k + 2] + kv.w * q_s[k + 3];
  }
  if (mask[(size_t)b * S_ + s]) sc = -INFINITY;
  red[tid] = sc; __syncthreads();
  for (int off = 128; off > 0; off >>= 1) {
    if (tid < off) red[tid] = fmaxf(red[tid], red[tid + off]);
    __syncthreads();
  }
  float mx = red[0]; __syncthreads();
  float e = __expf(sc - mx);
  red[tid] = e; __syncthreads();
  for (int off = 128; off > 0; off >>= 1) {
    if (tid < off) red[tid] += red[tid + off];
    __syncthreads();
  }
  float p = e / red[0];
  p_s[s] = p;
  attn_out[((size_t)t * B_ + b) * S_ + s] = p;
  __syncthreads();
  for (int n = tid; n < H_; n += 256) {
    float a = 0.f;
    const float* kb = att_keys + (size_t)b * S_ * H_ + n;
    for (int s2 = 0; s2 < S_; ++s2) a += p_s[s2] * kb[(size_t)s2 * H_];
    cat_bf[(size_t)b * (2 * H_) + n] = f2bf(a);  // attn_ctx half of concat (bf16)
    ctx_out[((size_t)t * B_ + b) * H_ + n] = a;  // fp32 output
  }
}

// ------- h_tilde = tanh(cat @ Wo^T + bo); 16x32 out per wave; writes output, h_bf, h_cur -------
__global__ __launch_bounds__(256) void out_gemm_tanh(const __bf16* __restrict__ cat_bf, // [B,2H]
                                                     const __bf16* __restrict__ Wo,    // [H,2H] bf16
                                                     const float* __restrict__ bo,
                                                     float* __restrict__ out_output,   // [B,T,H]
                                                     __bf16* __restrict__ h_bf,        // [B,H] bf16
                                                     float* __restrict__ h_cur,        // [B,H] fp32
                                                     int t) {
  int wave = threadIdx.x >> 5;
  int tile = blockIdx.x * (blockDim.x >> 5) + wave;
  const int tN = H_ >> 5;                        // 24 (2 n-tiles per wave)
  if (tile >= (B_ / 16) * tN) return;            // 96 waves
  int m0 = (tile / tN) << 4;
  int n0 = (tile % tN) << 5;
  int lane = threadIdx.x & 31;
  int l = lane & 15, half = lane >> 4;
  const int K = 2 * H_;
  const __bf16* arow = cat_bf + (size_t)(m0 + l) * K;
  const __bf16* w0 = Wo + (size_t)(n0 + l) * K;
  const __bf16* w1 = Wo + (size_t)(n0 + 16 + l) * K;
  v8f acc0 = {0.f, 0.f, 0.f, 0.f, 0.f, 0.f, 0.f, 0.f};
  v8f acc1 = acc0;
  for (int k0 = 0; k0 < K; k0 += 32) {
    v16bf av = load_a_bf(arow, k0, half);
    v16bf b0 = *(const v16bf*)(w0 + k0 + 16 * half);
    v16bf b1 = *(const v16bf*)(w1 + k0 + 16 * half);
    acc0 = __builtin_amdgcn_wmma_f32_16x16x32_bf16(false, av, false, b0, (short)0, acc0,
                                                   false, false);
    acc1 = __builtin_amdgcn_wmma_f32_16x16x32_bf16(false, av, false, b1, (short)0, acc1,
                                                   false, false);
  }
  int na = n0 + l, nb = n0 + 16 + l;
  float ba = bo[na], bb = bo[nb];
  for (int r = 0; r < 8; ++r) {
    int b = m0 + r + 8 * half;
    float va = tanhf(acc0[r] + ba);
    float vb = tanhf(acc1[r] + bb);
    out_output[((size_t)b * T_ + t) * H_ + na] = va;
    out_output[((size_t)b * T_ + t) * H_ + nb] = vb;
    h_bf[(size_t)b * H_ + na] = f2bf(va);
    h_bf[(size_t)b * H_ + nb] = f2bf(vb);
    h_cur[(size_t)b * H_ + na] = va;
    h_cur[(size_t)b * H_ + nb] = vb;
  }
}

__global__ __launch_bounds__(256) void finalize_hc(const float* __restrict__ h_cur,
                                                   const float* __restrict__ c_cur,
                                                   float* __restrict__ h_last,
                                                   float* __restrict__ c_last) {
  int i = blockIdx.x * blockDim.x + threadIdx.x;
  if (i < B_ * H_) { h_last[i] = h_cur[i]; c_last[i] = c_cur[i]; }
}

extern "C" void kernel_launch(void* const* d_in, const int* in_sizes, int n_in,
                              void* d_out, int out_size, void* d_ws, size_t ws_size,
                              hipStream_t stream) {
  (void)in_sizes; (void)n_in; (void)out_size; (void)ws_size;
  const float* x    = (const float*)d_in[0];
  const float* h0   = (const float*)d_in[1];
  const float* c0   = (const float*)d_in[2];
  const float* ctx  = (const float*)d_in[3];
  const unsigned char* mask = (const unsigned char*)d_in[4];
  const float* W_ih = (const float*)d_in[5];
  const float* b_ih = (const float*)d_in[6];
  const float* W_hh = (const float*)d_in[7];
  const float* b_hh = (const float*)d_in[8];
  const float* Wq   = (const float*)d_in[9];
  const float* bq   = (const float*)d_in[10];
  const float* Wk   = (const float*)d_in[11];
  const float* bk   = (const float*)d_in[12];
  const float* Wo   = (const float*)d_in[13];
  const float* bo   = (const float*)d_in[14];

  float* out        = (float*)d_out;
  float* out_output = out;                               // [B,T,H]
  float* out_hlast  = out_output + (size_t)B_ * T_ * H_;
  float* out_clast  = out_hlast + (size_t)B_ * H_;
  float* out_attn   = out_clast + (size_t)B_ * H_;       // [T,B,S]
  float* out_actx   = out_attn + (size_t)T_ * B_ * S_;   // [T,B,H]
  float* out_raw    = out_actx + (size_t)T_ * B_ * H_;   // [T,B,H]

  char* ws = (char*)d_ws;
  size_t cur = 0;
  auto alloc = [&](size_t bytes) -> void* {
    void* p = (void*)(ws + cur);
    cur += (bytes + 255) & ~(size_t)255;
    return p;
  };
  __bf16* Wih_bf  = (__bf16*)alloc((size_t)G_ * I_ * 2);
  __bf16* Whh_bf  = (__bf16*)alloc((size_t)G_ * H_ * 2);
  __bf16* Wq_bf   = (__bf16*)alloc((size_t)H_ * H_ * 2);
  __bf16* Wk_bf   = (__bf16*)alloc((size_t)H_ * H_ * 2);
  __bf16* Wo_bf   = (__bf16*)alloc((size_t)H_ * 2 * H_ * 2);
  __bf16* x_bf    = (__bf16*)alloc((size_t)B_ * T_ * I_ * 2);  // 12.6 MB
  __bf16* ctx_bf  = (__bf16*)alloc((size_t)B_ * S_ * H_ * 2);  // 25 MB
  __bf16* h_bf    = (__bf16*)alloc((size_t)B_ * H_ * 2);
  __bf16* cat_bf  = (__bf16*)alloc((size_t)B_ * 2 * H_ * 2);
  float*  gates_x = (float*)alloc((size_t)B_ * T_ * G_ * 4);   // 100 MB, L2-resident
  float*  att_k   = (float*)alloc((size_t)B_ * S_ * H_ * 4);   // 50 MB, L2-resident
  float*  h_cur   = (float*)alloc((size_t)B_ * H_ * 4);
  float*  c_cur   = (float*)alloc((size_t)B_ * H_ * 4);
  float*  qbuf    = (float*)alloc((size_t)B_ * H_ * 4);

  dim3 blk(256);
  // one-time conversions (weights + activations)
  cvt_bf16<<<(G_ * I_ + 255) / 256, blk, 0, stream>>>(W_ih, Wih_bf, G_ * I_);
  cvt_bf16<<<(G_ * H_ + 255) / 256, blk, 0, stream>>>(W_hh, Whh_bf, G_ * H_);
  cvt_bf16<<<(H_ * H_ + 255) / 256, blk, 0, stream>>>(Wq, Wq_bf, H_ * H_);
  cvt_bf16<<<(H_ * H_ + 255) / 256, blk, 0, stream>>>(Wk, Wk_bf, H_ * H_);
  cvt_bf16<<<(H_ * 2 * H_ + 255) / 256, blk, 0, stream>>>(Wo, Wo_bf, H_ * 2 * H_);
  cvt_bf16<<<(B_ * T_ * I_ + 255) / 256, blk, 0, stream>>>(x, x_bf, B_ * T_ * I_);
  cvt_bf16<<<(B_ * S_ * H_ + 255) / 256, blk, 0, stream>>>(ctx, ctx_bf, B_ * S_ * H_);
  cvt_bf16<<<(B_ * H_ + 255) / 256, blk, 0, stream>>>(h0, h_bf, B_ * H_);

  // Off-critical-path precomputation (both LSTM biases folded into gates_x once)
  {
    int tiles = (B_ * T_ / 16) * (G_ / 32);   // 49152 waves
    gemm_bf_2n<<<tiles / 8, blk, 0, stream>>>(x_bf, I_, Wih_bf, b_ih, b_hh, gates_x,
                                              B_ * T_, G_, I_);
  }
  {
    int tiles = (B_ * S_ / 16) * (H_ / 32);   // 24576 waves
    gemm_bf_2n<<<tiles / 8, blk, 0, stream>>>(ctx_bf, H_, Wk_bf, bk, nullptr, att_k,
                                              B_ * S_, H_, H_);
  }
  hipMemcpyAsync(c_cur, c0, (size_t)B_ * H_ * 4, hipMemcpyDeviceToDevice, stream);

  // Sequential recurrence: everything it touches stays in L2
  for (int t = 0; t < T_; ++t) {
    lstm_step<<<24, blk, 0, stream>>>(h_bf, Whh_bf, gates_x, c_cur, cat_bf, out_raw, t);
    // q = hy @ Wq^T + bq ; hy lives in cat_bf's second half (lda = 2H)
    gemm_bf_2n<<<12, blk, 0, stream>>>(cat_bf + H_, 2 * H_, Wq_bf, bq, nullptr, qbuf,
                                       B_, H_, H_);
    attention_step<<<B_, blk, 0, stream>>>(att_k, qbuf, mask, cat_bf, out_attn,
                                           out_actx, t);
    out_gemm_tanh<<<12, blk, 0, stream>>>(cat_bf, Wo_bf, bo, out_output, h_bf, h_cur, t);
  }
  finalize_hc<<<(B_ * H_ + 255) / 256, blk, 0, stream>>>(h_cur, c_cur, out_hlast, out_clast);
}
